// TemporalLogicMachineDP_45938970198159
// MI455X (gfx1250) — matrix operand order repaired and checked by reading, not compiled, not run on hardware
//
#include <hip/hip_runtime.h>
#include <hip/hip_bf16.h>

#define B_   2
#define T_   512
#define D_   64
#define REL_ 128
#define H_   64

typedef _Float16 v16h __attribute__((ext_vector_type(16)));
typedef float    v8f  __attribute__((ext_vector_type(8)));

union Frag16 { v16h v; _Float16 h[16]; };

// current0[b,i,c] : c<64 -> max_{t>=i} f[b,t,c] ; c>=64 -> min_{t>=i} f[b,t,c-64]
__global__ void __launch_bounds__(128) init_cur0_kernel(const float* __restrict__ f,
                                                        float* __restrict__ cur0) {
    int b  = blockIdx.x;
    int c  = threadIdx.x;        // 0..127
    int ch = c & 63;
    bool isMin = (c >= 64);
    float run = isMin ? __builtin_inff() : -__builtin_inff();
    for (int i = T_ - 1; i >= 0; --i) {
        float v = f[(b * T_ + i) * D_ + ch];
        run = isMin ? fminf(run, v) : fmaxf(run, v);
        cur0[(b * T_ + i) * REL_ + c] = run;
    }
}

// term1[b,j,o] = bias[o] + sum_c current[b,j,c] * W[c,o]   (top Ccur rows of W)
__global__ void __launch_bounds__(64) term1_kernel(const float* __restrict__ cur, int Ccur,
                                                   const float* __restrict__ W,
                                                   const float* __restrict__ bias,
                                                   float* __restrict__ t1) {
    int bj = blockIdx.x;         // b*T + j
    int o  = threadIdx.x;        // 0..63
    const float* cr = cur + bj * Ccur;
    float acc = bias[o];
    for (int c = 0; c < Ccur; ++c) acc += cr[c] * W[c * H_ + o];
    t1[bj * H_ + o] = acc;
}

// One block per (b,i). Regenerates r tiles in LDS, WMMA f16 GEMM against Wbot
// (bottom 128 rows of W), adds term1, fused backward max over j>=i of the
// pre-activation (sigmoid is monotonic -> applied once at the end).
__global__ void __launch_bounds__(128) main_kernel(const float* __restrict__ f,
                                                   const float* __restrict__ t1,
                                                   const float* __restrict__ Wbot,
                                                   float* __restrict__ out) {
    __shared__ _Float16 tileR[16 * 128];   // [j within tile][c]

    const int b    = blockIdx.x / T_;
    const int i    = blockIdx.x % T_;
    const int tid  = threadIdx.x;
    const int wave = tid >> 5;             // 0..3  -> output N tile
    const int lane = tid & 31;
    const int lo   = lane & 15;
    const int hi   = lane >> 4;
    const float NEG = -__builtin_inff();

    // ---- Preload B fragments (Wbot: [128,64] row-major f32 -> f16) ----
    // 32x16 f16 B fragment, this code's contiguous k order: lane holds column N = wave*16 + lo, K = h + 16*hi.
    Frag16 bf[4];
#pragma unroll
    for (int kk = 0; kk < 4; ++kk) {
        const int kbase = kk * 32;
        const int N = wave * 16 + lo;
#pragma unroll
        for (int h = 0; h < 16; ++h) {
            int K = kbase + h + hi * 16;
            bf[kk].h[h] = (_Float16)Wbot[K * 64 + N];
        }
    }

    // ---- Running interval-pool state: thread tid owns channel c ----
    const int  c     = tid;       // 0..127
    const int  ch    = c & 63;
    const bool isMin = (c >= 64);
    float run = isMin ? __builtin_inff() : NEG;

    float m = NEG;                // per-lane backward max of pre-activation
    const int jt0 = i >> 4;
    const int o   = wave * 16 + lo;

    for (int jt = jt0; jt < (T_ / 16); ++jt) {
        const int jbase = jt * 16;

        // Prefetch 16 independent f values, then branchless register scan.
        float vals[16];
#pragma unroll
        for (int jj = 0; jj < 16; ++jj) {
            vals[jj] = f[(b * T_ + jbase + jj) * D_ + ch];
        }
#pragma unroll
        for (int jj = 0; jj < 16; ++jj) {
            int j = jbase + jj;
            float nrun = isMin ? fminf(run, vals[jj]) : fmaxf(run, vals[jj]);
            run = (j >= i) ? nrun : run;
            tileR[jj * 128 + c] = (_Float16)((j >= i) ? run : 0.0f);
        }
        __syncthreads();

        // GEMM: [16 j x 128 k] x [128 k x 16 o], K in 4 steps of 32.
        v8f acc = {};
#pragma unroll
        for (int kk = 0; kk < 4; ++kk) {
            const int kbase = kk * 32;
            Frag16 af;
            // 16x32 f16 A fragment, contiguous k order used here: row M = lo; K = h + 16*hi (matches B).
#pragma unroll
            for (int h = 0; h < 16; ++h) {
                int K = h + hi * 16;
                af.h[h] = tileR[lo * 128 + kbase + K];
            }
            acc = __builtin_amdgcn_wmma_f32_16x16x32_f16(
                false, af.v, false, bf[kk].v, (short)0, acc, false, false);
        }

        // Epilogue: branchless masked max of (acc + term1) over j>=i.
        const float* t1p = t1 + (size_t)(b * T_ + jbase + hi * 8) * H_ + o;
        float tv[8];
#pragma unroll
        for (int r = 0; r < 8; ++r) tv[r] = t1p[r * H_];   // always in-bounds (j < T)
#pragma unroll
        for (int r = 0; r < 8; ++r) {
            int j = jbase + r + hi * 8;     // f32 C/D layout: M = r + 8*hi
            float val = acc[r] + tv[r];
            m = fmaxf(m, (j >= i) ? val : NEG);
        }
        __syncthreads();
    }

    // Lane L and L+16 saw complementary j rows for the same o: combine,
    // then apply sigmoid once (monotone, commutes with max).
    float other = __shfl_xor(m, 16, 32);
    m = fmaxf(m, other);
    if (hi == 0) {
        float s = __builtin_amdgcn_rcpf(1.0f + __expf(-m));
        out[(b * T_ + i) * H_ + o] = s;
    }
}

extern "C" void kernel_launch(void* const* d_in, const int* in_sizes, int n_in,
                              void* d_out, int out_size, void* d_ws, size_t ws_size,
                              hipStream_t stream) {
    const float* f  = (const float*)d_in[0];
    const float* W0 = (const float*)d_in[1];
    const float* b0 = (const float*)d_in[2];
    const float* W1 = (const float*)d_in[3];
    const float* b1 = (const float*)d_in[4];
    const float* W2 = (const float*)d_in[5];
    const float* b2 = (const float*)d_in[6];
    float* out = (float*)d_out;

    float* cur0 = (float*)d_ws;                 // B*T*128
    float* t1   = cur0 + B_ * T_ * REL_;        // B*T*64
    float* curA = t1   + B_ * T_ * H_;          // B*T*64
    float* curB = curA + B_ * T_ * H_;          // B*T*64

    init_cur0_kernel<<<B_, 128, 0, stream>>>(f, cur0);

    // Layer 0: Ccur = 128, W0 is [256,64]
    term1_kernel<<<B_ * T_, 64, 0, stream>>>(cur0, REL_, W0, b0, t1);
    main_kernel<<<B_ * T_, 128, 0, stream>>>(f, t1, W0 + REL_ * H_, curA);

    // Layer 1: Ccur = 64, W1 is [192,64]
    term1_kernel<<<B_ * T_, 64, 0, stream>>>(curA, H_, W1, b1, t1);
    main_kernel<<<B_ * T_, 128, 0, stream>>>(f, t1, W1 + H_ * H_, curB);

    // Layer 2: Ccur = 64, W2 is [192,64] -> final output
    term1_kernel<<<B_ * T_, 64, 0, stream>>>(curB, H_, W2, b2, t1);
    main_kernel<<<B_ * T_, 128, 0, stream>>>(f, t1, W2 + H_ * H_, out);
}
